// FusedMoE_12214886989881
// MI455X (gfx1250) — matrix operand order repaired and checked by reading, not compiled, not run on hardware
//
#include <hip/hip_runtime.h>
#include <hip/hip_bf16.h>
#include <math.h>

// ---------------------------------------------------------------------------
// Fused MoE for MI455X (gfx1250, wave32, WMMA bf16 -> f32 accumulate)
//   T=2048 H=1024 F=1024 E=8 K=2
// Pipeline: route -> convert x to bf16 -> GEMM1(gate_up)+SiLU -> GEMM2(down)
//           -> weighted gather (no atomics on output, deterministic).
// GEMM tiling: block 64x128, wave 32x32, TK=64.
// A-tiles staged with global_load_async_to_lds_b128 (ASYNCcnt path);
// weight tiles staged via load + v_cvt_pk_bf16_f32 + ds_store.
// Workspace usage ~34 MB.
// ---------------------------------------------------------------------------

#define T_  2048
#define H_  1024
#define F_  1024
#define E_  8
#define K_  2
#define CAP (T_ * K_)          // 4096 total pairs
#define ROWS (CAP + E_ * 64)   // padded rows (each expert padded to 64)

#define TM 64                  // block rows
#define TN 128                 // block cols
#define TK 64                  // k-step

// workspace byte offsets
#define OFF_META 0                     // 64 ints: [0..7]=counts [8..15]=cursor [16..24]=offs
#define OFF_PT   1024                  // pair_token: ROWS ints
#define OFF_PS   (OFF_PT + ROWS * 4)   // pair_slot:  T_*K_ ints
#define OFF_XB   (64 * 1024)           // x bf16: T_*H_*2 = 4 MB
#define OFF_ACT  (5  * 1024 * 1024)    // act bf16: ROWS*F_*2 ~= 9 MB
#define OFF_Y    (16 * 1024 * 1024)    // y f32:   ROWS*H_*4 ~= 18 MB

typedef __attribute__((ext_vector_type(16))) __bf16 v16bf;
typedef __attribute__((ext_vector_type(8)))  __bf16 v8bf;
typedef __attribute__((ext_vector_type(4)))  __bf16 v4bf;
typedef __attribute__((ext_vector_type(8)))  float  v8f;

// ---- CDNA5 async global->LDS helpers (ASYNCcnt-tracked) --------------------
// VDST is the 32-bit LDS byte address; generic shared pointers map to LDS via
// addr[31:0] (ISA 10.2 aperture rules), so truncation gives the LDS address.
__device__ __forceinline__ void async_ld_b128(void* lds_ptr, const void* gptr) {
    unsigned           lds  = (unsigned)(unsigned long long)lds_ptr;
    unsigned long long addr = (unsigned long long)gptr;
    asm volatile("global_load_async_to_lds_b128 %0, %1, off"
                 :: "v"(lds), "v"(addr) : "memory");
}
__device__ __forceinline__ void async_wait0() {
    asm volatile("s_wait_asynccnt 0x0" ::: "memory");
}

// ---------------- routing ----------------

__global__ void moe_zero_kernel(int* meta) {
    if (threadIdx.x < 64) meta[threadIdx.x] = 0;
}

__global__ void moe_count_kernel(const int* __restrict__ ids, int* __restrict__ meta) {
    int i = blockIdx.x * blockDim.x + threadIdx.x;
    if (i < T_ * K_) atomicAdd(&meta[ids[i]], 1);
}

__global__ void moe_scan_kernel(int* meta) {
    if (threadIdx.x == 0) {
        int off = 0;
        for (int e = 0; e < E_; ++e) {
            meta[16 + e] = off;
            off += ((meta[e] + TM - 1) / TM) * TM;   // pad each expert to 64 rows
        }
        meta[16 + E_] = off;
    }
}

__global__ void moe_assign_kernel(const int* __restrict__ ids, int* __restrict__ meta,
                                  int* __restrict__ pair_token, int* __restrict__ pair_slot) {
    int i = blockIdx.x * blockDim.x + threadIdx.x;
    if (i >= T_ * K_) return;
    int e = ids[i];
    int pos = atomicAdd(&meta[8 + e], 1);
    int slot = meta[16 + e] + pos;
    pair_token[slot] = i / K_;
    pair_slot[i] = slot;
}

// ---------------- f32 -> bf16 convert of hidden_states ----------------

__global__ void moe_cvt_x_kernel(const float* __restrict__ x, __bf16* __restrict__ xb) {
    int i = blockIdx.x * blockDim.x + threadIdx.x;     // T_*H_/4 threads
    const float4 v = *(const float4*)(x + (size_t)i * 4);
    v4bf o;
    o[0] = (__bf16)v.x; o[1] = (__bf16)v.y; o[2] = (__bf16)v.z; o[3] = (__bf16)v.w;
    *(v4bf*)(xb + (size_t)i * 4) = o;
}

// ---------------- WMMA fragment helpers (contiguous k = 16*half + s order, same for A and B) --------
// Tiles stored row-major in LDS with row stride TK (=64) bf16 elements.
// A 16x32 bf16: lane<16 holds M=lane, K=0..15; lane>=16 holds M=lane-16, K=16..31
// B 32x16 bf16: lane<16 holds N=lane, K=0..15; lane>=16 holds N=lane-16, K=16..31

__device__ __forceinline__ v16bf load_a_frag(const __bf16* A, int row, int ks, int half) {
    const __bf16* p = A + row * TK + ks * 32 + half * 16;
    v8bf lo = *(const v8bf*)(p);
    v8bf hi = *(const v8bf*)(p + 8);
    return __builtin_shufflevector(lo, hi, 0, 1, 2, 3, 4, 5, 6, 7,
                                           8, 9, 10, 11, 12, 13, 14, 15);
}

__device__ __forceinline__ v16bf load_b_frag(const __bf16* B, int n, int ks, int half) {
    return *(const v16bf*)(B + n * TK + ks * 32 + half * 16);
}

#define WMMA_BF16(A, B, C) __builtin_amdgcn_wmma_f32_16x16x32_bf16( \
        false, (A), false, (B), (short)0, (C), false, false)

// ---------------- GEMM1: act = silu(x@Wg^T) * (x@Wu^T), bf16 out ------------

__global__ __launch_bounds__(256) void moe_gemm1_kernel(
    const __bf16* __restrict__ xb,      // [T_][H_]
    const float*  __restrict__ wgu,     // [E_][2F_][H_]
    const int*    __restrict__ meta,
    const int*    __restrict__ pair_token,
    __bf16*       __restrict__ act)     // [ROWS][F_]
{
    const int e   = blockIdx.z;
    const int cnt = meta[e];
    const int mt  = blockIdx.y;
    if (mt * TM >= cnt) return;                  // block-uniform: EXEC stays full
    const int base = meta[16 + e];
    const int f0   = blockIdx.x * TN;

    __shared__ __align__(32) __bf16 sA [TM * TK];   //  8 KB
    __shared__ __align__(32) __bf16 sBg[TN * TK];   // 16 KB
    __shared__ __align__(32) __bf16 sBu[TN * TK];   // 16 KB

    const int tid  = threadIdx.x;
    const int lane = tid & 31;
    const int wid  = tid >> 5;
    const int wm   = wid >> 2;       // 0..1  -> 32-row slice
    const int wn   = wid & 3;        // 0..3  -> 32-col slice
    const int half = lane >> 4;
    const int r    = lane & 15;

    // A staging: two 16B chunks per thread (64 rows x 8 chunks)
    const int a_row = tid >> 2;                  // 0..63
    const int a_ch  = (tid & 3) * 2;             // chunk 0..7 (8 bf16 each)
    int srow = mt * TM + a_row;
    if (srow >= cnt) srow = cnt - 1;             // clamp pad rows
    const __bf16* xrow = xb + (size_t)pair_token[base + srow] * H_;
    __bf16* sA0 = &sA[a_row * TK + a_ch * 8];
    __bf16* sA1 = sA0 + 8;

    // B staging: 8 float4 per thread per matrix (128 rows x 16 float4)
    const int b_row = tid >> 1;                  // 0..127
    const int b_cb  = (tid & 1) * 8;             // float4 col base
    const size_t wbase = (size_t)e * (2 * F_) * H_;
    const float* gprow = wgu + wbase + (size_t)(f0 + b_row) * H_ + b_cb * 4;
    const float* uprow = gprow + (size_t)F_ * H_;

    v8f accg[2][2] = {};
    v8f accu[2][2] = {};

    for (int kk = 0; kk < H_ / TK; ++kk) {
        const int k0 = kk * TK;
        __syncthreads();
        // stage A via async global->LDS (bf16, no conversion needed)
        async_ld_b128(sA0, xrow + k0 + a_ch * 8);
        async_ld_b128(sA1, xrow + k0 + a_ch * 8 + 8);
        // stage B gate & up: load f32, convert to bf16, ds_store
#pragma unroll
        for (int q = 0; q < 8; ++q) {
            const float4 gv = *(const float4*)(gprow + k0 + q * 4);
            const float4 uv = *(const float4*)(uprow + k0 + q * 4);
            v4bf g; g[0] = (__bf16)gv.x; g[1] = (__bf16)gv.y; g[2] = (__bf16)gv.z; g[3] = (__bf16)gv.w;
            v4bf u; u[0] = (__bf16)uv.x; u[1] = (__bf16)uv.y; u[2] = (__bf16)uv.z; u[3] = (__bf16)uv.w;
            *(v4bf*)&sBg[b_row * TK + (b_cb + q) * 4] = g;
            *(v4bf*)&sBu[b_row * TK + (b_cb + q) * 4] = u;
        }
        if (kk + 1 < H_ / TK) {
            __builtin_prefetch(gprow + k0 + TK, 0, 1);   // global_prefetch_b8
            __builtin_prefetch(uprow + k0 + TK, 0, 1);
        }
        async_wait0();                // drain ASYNCcnt before barrier
        __syncthreads();

#pragma unroll
        for (int ks = 0; ks < 2; ++ks) {
            const v16bf a0 = load_a_frag(sA, wm * 32 + r,      ks, half);
            const v16bf a1 = load_a_frag(sA, wm * 32 + 16 + r, ks, half);
#pragma unroll
            for (int sn = 0; sn < 2; ++sn) {
                const v16bf bg = load_b_frag(sBg, wn * 32 + sn * 16 + r, ks, half);
                const v16bf bu = load_b_frag(sBu, wn * 32 + sn * 16 + r, ks, half);
                accg[0][sn] = WMMA_BF16(a0, bg, accg[0][sn]);
                accg[1][sn] = WMMA_BF16(a1, bg, accg[1][sn]);
                accu[0][sn] = WMMA_BF16(a0, bu, accu[0][sn]);
                accu[1][sn] = WMMA_BF16(a1, bu, accu[1][sn]);
            }
        }
    }

    // fused SiLU(gate) * up epilogue, bf16 out
    const size_t row0 = (size_t)base + (size_t)mt * TM;
#pragma unroll
    for (int sm = 0; sm < 2; ++sm) {
#pragma unroll
        for (int sn = 0; sn < 2; ++sn) {
            const int f = f0 + wn * 32 + sn * 16 + r;
#pragma unroll
            for (int j = 0; j < 8; ++j) {
                const int m = wm * 32 + sm * 16 + half * 8 + j;
                const float g = accg[sm][sn][j];
                const float u = accu[sm][sn][j];
                const float s = g / (1.0f + __expf(-g));
                act[(row0 + m) * F_ + f] = (__bf16)(s * u);
            }
        }
    }
}

// ---------------- GEMM2: y = act @ Wd^T, f32 out ----------------

__global__ __launch_bounds__(256) void moe_gemm2_kernel(
    const __bf16* __restrict__ act,     // [ROWS][F_]
    const float*  __restrict__ wd,      // [E_][H_][F_]
    const int*    __restrict__ meta,
    float*        __restrict__ ybuf)    // [ROWS][H_]
{
    const int e   = blockIdx.z;
    const int cnt = meta[e];
    const int mt  = blockIdx.y;
    if (mt * TM >= cnt) return;
    const int base = meta[16 + e];
    const int h0   = blockIdx.x * TN;

    __shared__ __align__(32) __bf16 sA[TM * TK];    //  8 KB
    __shared__ __align__(32) __bf16 sB[TN * TK];    // 16 KB

    const int tid  = threadIdx.x;
    const int lane = tid & 31;
    const int wid  = tid >> 5;
    const int wm   = wid >> 2;
    const int wn   = wid & 3;
    const int half = lane >> 4;
    const int r    = lane & 15;

    const size_t row0 = (size_t)base + (size_t)mt * TM;
    const int a_row = tid >> 2;
    const int a_ch  = (tid & 3) * 2;
    const __bf16* arow_p = act + (row0 + a_row) * F_;
    __bf16* sA0 = &sA[a_row * TK + a_ch * 8];
    __bf16* sA1 = sA0 + 8;

    const int b_row = tid >> 1;
    const int b_cb  = (tid & 1) * 8;
    const float* brow_p = wd + (size_t)e * H_ * F_ + (size_t)(h0 + b_row) * F_ + b_cb * 4;

    v8f acc[2][2] = {};

    for (int kk = 0; kk < F_ / TK; ++kk) {
        const int k0 = kk * TK;
        __syncthreads();
        async_ld_b128(sA0, arow_p + k0 + a_ch * 8);
        async_ld_b128(sA1, arow_p + k0 + a_ch * 8 + 8);
#pragma unroll
        for (int q = 0; q < 8; ++q) {
            const float4 v = *(const float4*)(brow_p + k0 + q * 4);
            v4bf b; b[0] = (__bf16)v.x; b[1] = (__bf16)v.y; b[2] = (__bf16)v.z; b[3] = (__bf16)v.w;
            *(v4bf*)&sB[b_row * TK + (b_cb + q) * 4] = b;
        }
        if (kk + 1 < F_ / TK) __builtin_prefetch(brow_p + k0 + TK, 0, 1);
        async_wait0();
        __syncthreads();

#pragma unroll
        for (int ks = 0; ks < 2; ++ks) {
            const v16bf a0 = load_a_frag(sA, wm * 32 + r,      ks, half);
            const v16bf a1 = load_a_frag(sA, wm * 32 + 16 + r, ks, half);
#pragma unroll
            for (int sn = 0; sn < 2; ++sn) {
                const v16bf b = load_b_frag(sB, wn * 32 + sn * 16 + r, ks, half);
                acc[0][sn] = WMMA_BF16(a0, b, acc[0][sn]);
                acc[1][sn] = WMMA_BF16(a1, b, acc[1][sn]);
            }
        }
    }

#pragma unroll
    for (int sm = 0; sm < 2; ++sm) {
#pragma unroll
        for (int sn = 0; sn < 2; ++sn) {
            const int h = h0 + wn * 32 + sn * 16 + r;
#pragma unroll
            for (int j = 0; j < 8; ++j) {
                const int m = wm * 32 + sm * 16 + half * 8 + j;
                ybuf[(row0 + m) * H_ + h] = acc[sm][sn][j];
            }
        }
    }
}

// ---------------- final gather: out[t] = sum_k w[t,k] * y[slot(t,k)] --------

__global__ void moe_gather_kernel(const float* __restrict__ ybuf,
                                  const float* __restrict__ tw,
                                  const int*   __restrict__ pair_slot,
                                  float*       __restrict__ out) {
    int i = blockIdx.x * blockDim.x + threadIdx.x;   // T_*H_/4 threads
    const int t = i / (H_ / 4);
    const int c = i % (H_ / 4);
    const int s0 = pair_slot[t * K_ + 0];
    const int s1 = pair_slot[t * K_ + 1];
    const float w0 = tw[t * K_ + 0];
    const float w1 = tw[t * K_ + 1];
    const float4 y0 = *(const float4*)(ybuf + (size_t)s0 * H_ + c * 4);
    const float4 y1 = *(const float4*)(ybuf + (size_t)s1 * H_ + c * 4);
    float4 o;
    o.x = w0 * y0.x + w1 * y1.x;
    o.y = w0 * y0.y + w1 * y1.y;
    o.z = w0 * y0.z + w1 * y1.z;
    o.w = w0 * y0.w + w1 * y1.w;
    *(float4*)(out + (size_t)t * H_ + c * 4) = o;
}

// ---------------- launch ----------------

extern "C" void kernel_launch(void* const* d_in, const int* in_sizes, int n_in,
                              void* d_out, int out_size, void* d_ws, size_t ws_size,
                              hipStream_t stream) {
    (void)in_sizes; (void)n_in; (void)out_size; (void)ws_size;
    const float* x   = (const float*)d_in[0];   // hidden_states [T,H]
    const float* tw  = (const float*)d_in[1];   // topk_weights  [T,K]
    const int*   ids = (const int*)  d_in[2];   // topk_ids      [T,K]
    const float* wgu = (const float*)d_in[3];   // gate_up_weight[E,2F,H]
    const float* wd  = (const float*)d_in[4];   // down_weight   [E,H,F]
    float* out = (float*)d_out;

    char* ws = (char*)d_ws;
    int*    meta       = (int*)   (ws + OFF_META);
    int*    pair_token = (int*)   (ws + OFF_PT);
    int*    pair_slot  = (int*)   (ws + OFF_PS);
    __bf16* xb         = (__bf16*)(ws + OFF_XB);
    __bf16* act        = (__bf16*)(ws + OFF_ACT);
    float*  ybuf       = (float*) (ws + OFF_Y);

    moe_zero_kernel  <<<1, 64, 0, stream>>>(meta);
    moe_count_kernel <<<(T_ * K_) / 256, 256, 0, stream>>>(ids, meta);
    moe_scan_kernel  <<<1, 32, 0, stream>>>(meta);
    moe_assign_kernel<<<(T_ * K_) / 256, 256, 0, stream>>>(ids, meta, pair_token, pair_slot);
    moe_cvt_x_kernel <<<(T_ * H_ / 4) / 256, 256, 0, stream>>>(x, xb);

    dim3 g1(F_ / TN, CAP / TM, E_);
    moe_gemm1_kernel<<<g1, 256, 0, stream>>>(xb, wgu, meta, pair_token, act);
    dim3 g2(H_ / TN, CAP / TM, E_);
    moe_gemm2_kernel<<<g2, 256, 0, stream>>>(act, wd, meta, ybuf);

    moe_gather_kernel<<<(T_ * H_ / 4) / 256, 256, 0, stream>>>(ybuf, tw, pair_slot, out);
}
